// MicrotubuleAttention_21431886807331
// MI455X (gfx1250) — matrix operand order repaired and checked by reading, not compiled, not run on hardware
//
#include <hip/hip_runtime.h>
#include <math.h>

typedef __attribute__((ext_vector_type(16))) __bf16 v16bf;
typedef __attribute__((ext_vector_type(8)))  __bf16 v8bf;
typedef __attribute__((ext_vector_type(8)))  float  v8f;

#define B_    2
#define T_    2048
#define DM    1024
#define HQ    16
#define HKV   4
#define DH    64
#define NEGV  -1000000000.0f

static __device__ __forceinline__ unsigned short f2bf(float f) {
  unsigned int u = __float_as_uint(f);
  u += 0x7FFFu + ((u >> 16) & 1u);          // round-to-nearest-even
  return (unsigned short)(u >> 16);
}

static __device__ __forceinline__ v16bf ld16(const unsigned short* p0,
                                             const unsigned short* p1) {
  v8bf lo = *reinterpret_cast<const v8bf*>(p0);
  v8bf hi = *reinterpret_cast<const v8bf*>(p1);
  return __builtin_shufflevector(lo, hi, 0,1,2,3,4,5,6,7,8,9,10,11,12,13,14,15);
}

static __device__ __forceinline__ v8f wmma_bf16(v16bf a, v16bf b, v8f c) {
  return __builtin_amdgcn_wmma_f32_16x16x32_bf16(false, a, false, b,
                                                 (short)0, c, false, false);
}

// async copy 16B/lane global -> LDS, tracked by ASYNCcnt
static __device__ __forceinline__ void async_b128(unsigned lds_off,
                                                  const void* gptr) {
  asm volatile("global_load_async_to_lds_b128 %0, %1, off"
               :: "v"(lds_off), "v"(gptr) : "memory");
}

// ---------------------------------------------------------------- conversions
__global__ void k_f32_to_bf16(const float* __restrict__ src,
                              unsigned short* __restrict__ dst, int n) {
  int i = blockIdx.x * blockDim.x + threadIdx.x;
  if (i < n) dst[i] = f2bf(src[i]);
}

// src: K x N (f32, row-major) -> dst: N x K (bf16, row-major)
__global__ void k_transpose_to_bf16(const float* __restrict__ src,
                                    unsigned short* __restrict__ dst,
                                    int K, int N) {
  int i = blockIdx.x * blockDim.x + threadIdx.x;
  if (i >= K * N) return;
  int k = i / N, n = i - k * N;
  dst[(size_t)n * K + k] = f2bf(src[i]);
}

// ---------------------------------------------------------------- bf16 GEMM
// C(MxN,f32) = A(MxK,bf16 row-major) * Bt(NxK,bf16 row-major)^T
__global__ __launch_bounds__(32)
void k_gemm_bf16(const unsigned short* __restrict__ A,
                 const unsigned short* __restrict__ Bt,
                 float* __restrict__ C, int M, int N, int K) {
  const int lane = threadIdx.x;
  const int hf = lane >> 4, ln = lane & 15;
  const int m0 = blockIdx.x * 32, n0 = blockIdx.y * 32;
  v8f acc00 = {}, acc01 = {}, acc10 = {}, acc11 = {};
  const unsigned short* arow0 = A  + (size_t)(m0 + ln)      * K;
  const unsigned short* arow1 = A  + (size_t)(m0 + 16 + ln) * K;
  const unsigned short* brow0 = Bt + (size_t)(n0 + ln)      * K;
  const unsigned short* brow1 = Bt + (size_t)(n0 + 16 + ln) * K;
  for (int kk = 0; kk < K; kk += 32) {
    __builtin_prefetch(arow0 + kk + 128, 0, 3);
    __builtin_prefetch(brow0 + kk + 128, 0, 3);
    const int ao = kk + hf * 8;      // A 16x32: halves hold K 0-7/16-23 vs 8-15/24-31
    v16bf a0 = ld16(arow0 + ao, arow0 + ao + 16);
    v16bf a1 = ld16(arow1 + ao, arow1 + ao + 16);
    const int bo = kk + hf * 8;      // B 32x16: halves hold K 0-7/16-23 vs 8-15/24-31
    v16bf b0 = ld16(brow0 + bo, brow0 + bo + 16);
    v16bf b1 = ld16(brow1 + bo, brow1 + bo + 16);
    acc00 = wmma_bf16(a0, b0, acc00);
    acc01 = wmma_bf16(a0, b1, acc01);
    acc10 = wmma_bf16(a1, b0, acc10);
    acc11 = wmma_bf16(a1, b1, acc11);
  }
  const int ro = hf * 8;             // C layout: lanes 0-15 rows 0-7, 16-31 rows 8-15
  for (int r = 0; r < 8; ++r) {
    C[(size_t)(m0 + r + ro) * N + n0 + ln]           = acc00[r];
    C[(size_t)(m0 + r + ro) * N + n0 + 16 + ln]      = acc01[r];
    C[(size_t)(m0 + 16 + r + ro) * N + n0 + ln]      = acc10[r];
    C[(size_t)(m0 + 16 + r + ro) * N + n0 + 16 + ln] = acc11[r];
  }
}

// ---------------------------------------------------------------- RoPE / V
// qkv: [B*T][1536] f32 : cols 0..1023 Q, 1024..1279 K, 1280..1535 V
__global__ void k_rope_q(const float* __restrict__ qkv,
                         unsigned short* __restrict__ Qb) {
  int i = blockIdx.x * blockDim.x + threadIdx.x;     // B*HQ*T*32 = 2^21
  int d = i & 31, t = (i >> 5) & (T_ - 1), h = (i >> 16) & (HQ - 1), b = i >> 20;
  const float* row = qkv + (size_t)(b * T_ + t) * 1536 + h * 64;
  float x1 = row[d], x2 = row[d + 32];
  float inv = powf(10000.0f, -(float)d * (1.0f / 32.0f));
  float ang = (float)t * inv;
  float c = cosf(ang), s = sinf(ang);
  unsigned short* orow = Qb + (((size_t)(b * HQ + h) * T_) + t) * DH;
  orow[d]      = f2bf(x1 * c - x2 * s);
  orow[d + 32] = f2bf(x2 * c + x1 * s);
}

__global__ void k_rope_k(const float* __restrict__ qkv,
                         unsigned short* __restrict__ Kb) {
  int i = blockIdx.x * blockDim.x + threadIdx.x;     // B*HKV*T*32 = 2^19
  int d = i & 31, t = (i >> 5) & (T_ - 1), kv = (i >> 16) & (HKV - 1), b = i >> 18;
  const float* row = qkv + (size_t)(b * T_ + t) * 1536 + 1024 + kv * 64;
  float x1 = row[d], x2 = row[d + 32];
  float inv = powf(10000.0f, -(float)d * (1.0f / 32.0f));
  float ang = (float)t * inv;
  float c = cosf(ang), s = sinf(ang);
  unsigned short* orow = Kb + (((size_t)(b * HKV + kv) * T_) + t) * DH;
  orow[d]      = f2bf(x1 * c - x2 * s);
  orow[d + 32] = f2bf(x2 * c + x1 * s);
}

// V transposed to [B][HKV][DH][T] so PV B-operand reads are contiguous in t
__global__ void k_v_transpose(const float* __restrict__ qkv,
                              unsigned short* __restrict__ Vtb) {
  int i = blockIdx.x * blockDim.x + threadIdx.x;     // B*HKV*T*64 = 2^20
  int d = i & 63, t = (i >> 6) & (T_ - 1), kv = (i >> 17) & (HKV - 1), b = i >> 19;
  float v = qkv[(size_t)(b * T_ + t) * 1536 + 1280 + kv * 64 + d];
  Vtb[(((size_t)(b * HKV + kv) * DH) + d) * T_ + t] = f2bf(v);
}

// ---------------------------------------------------------------- flash attn
// one wave per (b, h, 32-query tile); 32-key blocks double-buffered through
// LDS with global_load_async_to_lds_b128 (ASYNCcnt), online softmax, WMMA.
static __device__ __forceinline__ void stage_kv(const unsigned short* Kg,
                                                const unsigned short* Vg,
                                                unsigned klds, unsigned vlds,
                                                int lane) {
#pragma unroll
  for (int j = 0; j < 8; ++j) {                 // K block: 32 keys x 64 d, row-major
    int chunk = j * 32 + lane;                  // 256 x 16B
    int key = chunk >> 3, doff = (chunk & 7) * 8;
    async_b128(klds + key * 128 + doff * 2, Kg + key * DH + doff);
  }
#pragma unroll
  for (int j = 0; j < 8; ++j) {                 // V block: 64 d x 32 keys, d-major
    int d = j * 8 + (lane >> 2), koff = (lane & 3) * 8;
    async_b128(vlds + d * 64 + koff * 2, Vg + (size_t)d * T_ + koff);
  }
}

__global__ __launch_bounds__(32)
void k_flash(const unsigned short* __restrict__ Qb,   // [B][HQ][T][DH]
             const unsigned short* __restrict__ Kb,   // [B][HKV][T][DH]
             const unsigned short* __restrict__ Vtb,  // [B][HKV][DH][T]
             const float* __restrict__ pol_dir,
             const float* __restrict__ gtp_gamma,
             unsigned short* __restrict__ Ob) {       // [B][T][HQ*DH]
  __shared__ __attribute__((aligned(16))) unsigned short Kbuf[2][32 * DH];
  __shared__ __attribute__((aligned(16))) unsigned short Vbuf[2][DH * 32];
  __shared__ __attribute__((aligned(16))) unsigned short Plds[32 * 40];

  const int lane = threadIdx.x;
  const int hf = lane >> 4, ln = lane & 15;
  int idx = blockIdx.x;
  const int qt = idx & (T_ / 32 - 1); idx >>= 6;
  const int h  = idx & (HQ - 1);      idx >>= 4;
  const int b  = idx;
  const int q0 = qt * 32;
  const int kv = h >> 2;                       // N_REP = 4

  const float pol   = fminf(1.f, fmaxf(-1.f, pol_dir[h]));
  const float gamma = fmaxf(1e-4f, gtp_gamma[h]);

  const unsigned short* Qbase = Qb  + (((size_t)(b * HQ  + h ) * T_) + q0) * DH;
  const unsigned short* Kbase = Kb  + ((size_t)(b * HKV + kv) * T_) * DH;
  const unsigned short* Vbase = Vtb + ((size_t)(b * HKV + kv) * DH) * T_;

  const unsigned kOff0 = (unsigned)(size_t)&Kbuf[0][0];
  const unsigned kOff1 = (unsigned)(size_t)&Kbuf[1][0];
  const unsigned vOff0 = (unsigned)(size_t)&Vbuf[0][0];
  const unsigned vOff1 = (unsigned)(size_t)&Vbuf[1][0];

  // Q A-operands: 2 row-tiles x 2 d-chunks
  v16bf aQ[2][2];
#pragma unroll
  for (int qa = 0; qa < 2; ++qa) {
    const unsigned short* qrow = Qbase + (size_t)(qa * 16 + ln) * DH;
    aQ[qa][0] = ld16(qrow + hf * 8,      qrow + 16 + hf * 8);
    aQ[qa][1] = ld16(qrow + 32 + hf * 8, qrow + 48 + hf * 8);
  }

  const v8f vzero = {};
  v8f O[2][4];
  float m_i[2][8], l_i[2][8];
#pragma unroll
  for (int qa = 0; qa < 2; ++qa) {
#pragma unroll
    for (int dt = 0; dt < 4; ++dt) O[qa][dt] = vzero;
#pragma unroll
    for (int r = 0; r < 8; ++r) { m_i[qa][r] = -3.0e38f; l_i[qa][r] = 0.f; }
  }

  const int nkb = qt + 1;                      // causal: skip fully-masked blocks
  stage_kv(Kbase, Vbase, kOff0, vOff0, lane);  // prologue fill of buffer 0

  for (int kb = 0; kb < nkb; ++kb) {
    const int ks = kb * 32;
    const int cur = kb & 1;
    if (kb + 1 < nkb) {                        // prefetch next block, other buffer
      stage_kv(Kbase + (size_t)(ks + 32) * DH, Vbase + ks + 32,
               cur ? kOff0 : kOff1, cur ? vOff0 : vOff1, lane);
      asm volatile("s_wait_asynccnt 0x10" ::: "memory");  // current buffer ready
    } else {
      asm volatile("s_wait_asynccnt 0x0" ::: "memory");
    }
    const unsigned short* Kl = &Kbuf[cur][0];
    const unsigned short* Vl = &Vbuf[cur][0];

    // ---- S = Q K^T from LDS
    v16bf bK[2][2];
#pragma unroll
    for (int nt = 0; nt < 2; ++nt) {
      const unsigned short* kr = Kl + (nt * 16 + ln) * DH;
      bK[nt][0] = ld16(kr + hf * 8,      kr + 16 + hf * 8);
      bK[nt][1] = ld16(kr + 32 + hf * 8, kr + 48 + hf * 8);
    }
    v8f S[2][2];
#pragma unroll
    for (int qa = 0; qa < 2; ++qa)
#pragma unroll
      for (int nt = 0; nt < 2; ++nt) {
        v8f s = vzero;
        s = wmma_bf16(aQ[qa][0], bK[nt][0], s);
        s = wmma_bf16(aQ[qa][1], bK[nt][1], s);
        S[qa][nt] = s;
      }

    // ---- bias + online softmax (C layout: row = r + 8*hf, col = ln / 16+ln)
#pragma unroll
    for (int qa = 0; qa < 2; ++qa) {
      float P0[8], P1[8];
#pragma unroll
      for (int r = 0; r < 8; ++r) {
        const int qpos = q0 + qa * 16 + r + hf * 8;
        float s0 = S[qa][0][r] * 0.125f;
        float s1 = S[qa][1][r] * 0.125f;
        const int k0p = ks + ln, k1p = ks + 16 + ln;
        if (k0p <= qpos) {
          float dd = (float)(qpos - k0p);
          s0 += pol * (-dd) * (1.0f / 2048.0f) + (1.0f - __expf(-gamma * dd)) * NEGV;
        } else s0 += NEGV;
        if (k1p <= qpos) {
          float dd = (float)(qpos - k1p);
          s1 += pol * (-dd) * (1.0f / 2048.0f) + (1.0f - __expf(-gamma * dd)) * NEGV;
        } else s1 += NEGV;

        float v = fmaxf(s0, s1);               // row max over 32 keys
        v = fmaxf(v, __shfl_xor(v, 1, 16));
        v = fmaxf(v, __shfl_xor(v, 2, 16));
        v = fmaxf(v, __shfl_xor(v, 4, 16));
        v = fmaxf(v, __shfl_xor(v, 8, 16));
        const float mnew  = fmaxf(m_i[qa][r], v);
        const float alpha = __expf(m_i[qa][r] - mnew);
        m_i[qa][r] = mnew;
        float p0 = __expf(s0 - mnew), p1 = __expf(s1 - mnew);
        float rs = p0 + p1;
        rs += __shfl_xor(rs, 1, 16);
        rs += __shfl_xor(rs, 2, 16);
        rs += __shfl_xor(rs, 4, 16);
        rs += __shfl_xor(rs, 8, 16);
        l_i[qa][r] = l_i[qa][r] * alpha + rs;
        P0[r] = p0; P1[r] = p1;
#pragma unroll
        for (int dt = 0; dt < 4; ++dt) O[qa][dt][r] = O[qa][dt][r] * alpha;
      }
#pragma unroll
      for (int r = 0; r < 8; ++r) {
        Plds[(qa * 16 + r + hf * 8) * 40 + ln]      = f2bf(P0[r]);
        Plds[(qa * 16 + r + hf * 8) * 40 + 16 + ln] = f2bf(P1[r]);
      }
    }
    __syncthreads();
    v16bf aP[2];
    aP[0] = ld16(&Plds[ln * 40 + hf * 8],        &Plds[ln * 40 + 16 + hf * 8]);
    aP[1] = ld16(&Plds[(16 + ln) * 40 + hf * 8], &Plds[(16 + ln) * 40 + 16 + hf * 8]);
    __syncthreads();

    // ---- O += P V from LDS (V d-major: [d][key])
#pragma unroll
    for (int dt = 0; dt < 4; ++dt) {
      const unsigned short* vr = Vl + (dt * 16 + ln) * 32 + hf * 8;
      v16bf bV = ld16(vr, vr + 16);
      O[0][dt] = wmma_bf16(aP[0], bV, O[0][dt]);
      O[1][dt] = wmma_bf16(aP[1], bV, O[1][dt]);
    }
  }

  // ---- normalize + store bf16 to [B][T][HQ*DH]
#pragma unroll
  for (int qa = 0; qa < 2; ++qa)
#pragma unroll
    for (int r = 0; r < 8; ++r) {
      const float inv = 1.0f / l_i[qa][r];
      const int t = q0 + qa * 16 + r + hf * 8;
      unsigned short* orow = Ob + ((size_t)(b * T_ + t) * (HQ * DH)) + h * DH;
#pragma unroll
      for (int dt = 0; dt < 4; ++dt)
        orow[dt * 16 + ln] = f2bf(O[qa][dt][r] * inv);
    }
}

// ---------------------------------------------------------------- launch
extern "C" void kernel_launch(void* const* d_in, const int* in_sizes, int n_in,
                              void* d_out, int out_size, void* d_ws, size_t ws_size,
                              hipStream_t stream) {
  const float* x   = (const float*)d_in[0];
  const float* Wq  = (const float*)d_in[1];
  const float* Wk  = (const float*)d_in[2];
  const float* Wv  = (const float*)d_in[3];
  const float* Wo  = (const float*)d_in[4];
  const float* pol = (const float*)d_in[5];
  const float* gam = (const float*)d_in[6];
  float* out = (float*)d_out;

  char* w = (char*)d_ws;
  unsigned short* xb    = (unsigned short*)w; w += (size_t)4096 * 1024 * 2; // x bf16
  unsigned short* wqkvT = (unsigned short*)w; w += (size_t)1536 * 1024 * 2; // [Wq;Wk;Wv]^T bf16
  unsigned short* woT   = (unsigned short*)w; w += (size_t)1024 * 1024 * 2; // Wo^T bf16
  float*          qkv   = (float*)w;          w += (size_t)4096 * 1536 * 4; // raw QKV f32
  unsigned short* Qb    = (unsigned short*)w; w += (size_t)B_ * HQ  * T_ * DH * 2;
  unsigned short* Kb    = (unsigned short*)w; w += (size_t)B_ * HKV * T_ * DH * 2;
  unsigned short* Vtb   = (unsigned short*)w; w += (size_t)B_ * HKV * T_ * DH * 2;
  unsigned short* Ab    = (unsigned short*)w; w += (size_t)4096 * 1024 * 2; // attn out bf16

  k_f32_to_bf16<<<4194304 / 256, 256, 0, stream>>>(x, xb, 4194304);
  k_transpose_to_bf16<<<1048576 / 256, 256, 0, stream>>>(Wq, wqkvT, 1024, 1024);
  k_transpose_to_bf16<<<262144 / 256, 256, 0, stream>>>(Wk, wqkvT + (size_t)1024 * 1024, 1024, 256);
  k_transpose_to_bf16<<<262144 / 256, 256, 0, stream>>>(Wv, wqkvT + (size_t)1280 * 1024, 1024, 256);
  k_transpose_to_bf16<<<1048576 / 256, 256, 0, stream>>>(Wo, woT, 1024, 1024);

  dim3 g1(4096 / 32, 1536 / 32);
  k_gemm_bf16<<<g1, 32, 0, stream>>>(xb, wqkvT, qkv, 4096, 1536, 1024);

  k_rope_q<<<2097152 / 256, 256, 0, stream>>>(qkv, Qb);
  k_rope_k<<<524288 / 256, 256, 0, stream>>>(qkv, Kb);
  k_v_transpose<<<1048576 / 256, 256, 0, stream>>>(qkv, Vtb);

  k_flash<<<B_ * HQ * (T_ / 32), 32, 0, stream>>>(Qb, Kb, Vtb, pol, gam, Ab);

  dim3 g2(4096 / 32, 1024 / 32);
  k_gemm_bf16<<<g2, 32, 0, stream>>>(Ab, woT, out, 4096, 1024, 1024);
}
